// patch_merging_layer_26388279067307
// MI455X (gfx1250) — matrix operand order repaired and checked by reading, not compiled, not run on hardware
//
#include <hip/hip_runtime.h>

typedef __attribute__((ext_vector_type(16))) __bf16 v16bf;
typedef __attribute__((ext_vector_type(8)))  __bf16 v8bf;
typedef __attribute__((ext_vector_type(8)))  float  v8f;

#define B_       8
#define F_       64
#define NIN_     40962
#define N_       10242
#define OUT_     128
#define K_       448          // 7*F
#define KSTEPS   14           // 448/32
#define NT_BLK   8            // n-tiles (of 16 cols) per block
#define BPB      81           // ceil(641 tiles / 8)
#define LDS_PITCH 464         // 448 + 16 pad -> 928B rows, 32B aligned

// sched_group_barrier masks
#define SGB_WMMA    0x008     // MFMA/WMMA class
#define SGB_DSREAD  0x100     // DS read

// ---------------------------------------------------------------------------
// Kernel 1: gather + GEMM (bf16 WMMA) + bias, writes y = W*g + b to d_out
// Double-buffered LDS tile, one barrier per tile, branch-free gather,
// distance-2 DS-read / WMMA software pipeline enforced via sched barriers.
// ---------------------------------------------------------------------------
__global__ __launch_bounds__(256) void fused_gather_gemm_kernel(
    const float* __restrict__ x, const long long* __restrict__ neigh,
    const float* __restrict__ W, const float* __restrict__ bias,
    float* __restrict__ y)
{
  __shared__ __bf16 gtile[2][16][LDS_PITCH];   // double-buffered [n_local][k]

  const int tid  = threadIdx.x;
  const int lane = tid & 31;
  const int wave = tid >> 5;        // 0..7 -> which 16 output channels
  const int half = lane >> 4;       // 0/1: half-wave
  const int l15  = lane & 15;       // A: M row / B,C: N col

  const int b     = blockIdx.x / BPB;
  const int tile0 = (blockIdx.x % BPB) * NT_BLK;
  const float* xb = x + (long long)b * F_ * NIN_;

  // ---- Preload A fragments (W rows as bf16, contiguous-K A order, same as B) ----
  // lane l: M = l&15; VGPR v covers K = (l>>4)*16 + v*2 (+1)
  v16bf A[KSTEPS];
  {
    __bf16* ae = (__bf16*)A;
    const float* wrow = W + (wave * 16 + l15) * K_;
#pragma unroll
    for (int s = 0; s < KSTEPS; ++s) {
#pragma unroll
      for (int v = 0; v < 8; ++v) {
        const int k = s * 32 + half * 16 + v * 2;
        const float2 wv = *(const float2*)(wrow + k);
        ae[s * 16 + 2 * v]     = (__bf16)wv.x;
        ae[s * 16 + 2 * v + 1] = (__bf16)wv.y;
      }
    }
  }
  float bv[8];
#pragma unroll
  for (int r = 0; r < 8; ++r) bv[r] = bias[wave * 16 + half * 8 + r];

  for (int t = 0; t < NT_BLK; ++t) {
    const int nbase = (tile0 + t) * 16;
    if (nbase >= N_) break;               // uniform across block
    const int buf = t & 1;

    // ---- gather 448x16 tile into LDS, branch-free (clamped indices) ----
    // p -> (n_local = p&15, k-block = (p>>4)*8); 8 f-rows packed per b128.
    for (int p = tid; p < 896; p += 256) {
      const int nl = p & 15;
      const int k  = (p >> 4) << 3;       // 0,8,...,440 (same j across 8)
      const int j  = k >> 6;
      const int f  = k & 63;
      int n = nbase + nl;
      n = (n < N_) ? n : (N_ - 1);        // clamp: junk cols masked at store
      const int col = (int)neigh[n * 7 + j];
      const float* src = xb + (long long)f * NIN_ + col;
      v8bf pk;
#pragma unroll
      for (int q = 0; q < 8; ++q)
        pk[q] = (__bf16)src[q * NIN_];
      *(v8bf*)&gtile[buf][nl][k] = pk;
    }
    __syncthreads();   // tile t written; t-2 readers of this buffer long done

    // ---- WMMA: C(16x16 f32) += A(16x32) * B(32x16), pipelined B loads ----
    // B lane layout: n = l&15, K-base = (l>>4)*16, 16 contiguous K per frag
    v8f C = {};
    const __bf16* grow = &gtile[buf][l15][half * 16];
    v16bf Bf[3];
    Bf[0] = *(const v16bf*)(grow);
    Bf[1] = *(const v16bf*)(grow + 32);
    Bf[2] = *(const v16bf*)(grow + 64);
#pragma unroll
    for (int s = 0; s < KSTEPS; ++s) {
      C = __builtin_amdgcn_wmma_f32_16x16x32_bf16(false, A[s], false, Bf[s % 3],
                                                  (short)0, C, false, false);
      if (s + 3 < KSTEPS)
        Bf[s % 3] = *(const v16bf*)(grow + (s + 3) * 32);
    }
    // Enforce distance-2 pipeline: 6 DS reads up front, then 1 WMMA : 2 DS
    // reads interleaved, final 3 WMMAs drain. (Hint only; no semantics.)
    __builtin_amdgcn_sched_group_barrier(SGB_DSREAD, 6, 0);
#pragma unroll
    for (int s = 0; s < KSTEPS - 3; ++s) {
      __builtin_amdgcn_sched_group_barrier(SGB_WMMA,   1, 0);
      __builtin_amdgcn_sched_group_barrier(SGB_DSREAD, 2, 0);
    }
    __builtin_amdgcn_sched_group_barrier(SGB_WMMA, 3, 0);

    // ---- epilogue: bias + store (C: m = half*8 + r, n = l&15) ----
    const int ncol = nbase + l15;
    if (ncol < N_) {
#pragma unroll
      for (int r = 0; r < 8; ++r) {
        const int o = wave * 16 + half * 8 + r;
        y[((long long)b * OUT_ + o) * N_ + ncol] = C[r] + bv[r];
      }
    }
  }
}

// ---------------------------------------------------------------------------
// Kernel 2: deterministic per-channel mean / inv-std (no float atomics)
// ---------------------------------------------------------------------------
__global__ __launch_bounds__(256) void bn_stats_kernel(
    const float* __restrict__ y, float* __restrict__ stats)
{
  __shared__ float rs[256], rs2[256];
  const int o = blockIdx.x, tid = threadIdx.x;
  float s = 0.0f, s2 = 0.0f;
  for (int bb = 0; bb < B_; ++bb) {
    const float* row = y + ((long long)bb * OUT_ + o) * N_;
    for (int n = tid; n < N_; n += 256) {
      const float v = row[n];
      s += v; s2 += v * v;
    }
  }
  rs[tid] = s; rs2[tid] = s2;
  __syncthreads();
  for (int st = 128; st > 0; st >>= 1) {
    if (tid < st) { rs[tid] += rs[tid + st]; rs2[tid] += rs2[tid + st]; }
    __syncthreads();
  }
  if (tid == 0) {
    const float cnt  = (float)B_ * (float)N_;
    const float mean = rs[0] / cnt;
    const float var  = rs2[0] / cnt - mean * mean;
    stats[o]         = mean;
    stats[OUT_ + o]  = rsqrtf(var + 1e-5f);
  }
}

// ---------------------------------------------------------------------------
// Kernel 3: in-place normalize + gamma/beta
// ---------------------------------------------------------------------------
__global__ __launch_bounds__(256) void bn_apply_kernel(
    float* __restrict__ y, const float* __restrict__ stats,
    const float* __restrict__ gamma, const float* __restrict__ beta)
{
  const int i = blockIdx.x * 256 + threadIdx.x;   // < 10,487,808
  const int o = (i / N_) & (OUT_ - 1);            // (i/N) % 128
  const float mean = stats[o];
  const float inv  = stats[OUT_ + o];
  y[i] = (y[i] - mean) * inv * gamma[o] + beta[o];
}

// ---------------------------------------------------------------------------
extern "C" void kernel_launch(void* const* d_in, const int* in_sizes, int n_in,
                              void* d_out, int out_size, void* d_ws, size_t ws_size,
                              hipStream_t stream) {
  const float*     x     = (const float*)d_in[0];
  const long long* neigh = (const long long*)d_in[1];
  const float*     W     = (const float*)d_in[2];
  const float*     bias  = (const float*)d_in[3];
  const float*     gamma = (const float*)d_in[4];
  const float*     beta  = (const float*)d_in[5];
  float* y     = (float*)d_out;
  float* stats = (float*)d_ws;                    // 256 floats

  fused_gather_gemm_kernel<<<B_ * BPB, 256, 0, stream>>>(x, neigh, W, bias, y);
  bn_stats_kernel<<<OUT_, 256, 0, stream>>>(y, stats);
  bn_apply_kernel<<<(B_ * OUT_ * N_) / 256, 256, 0, stream>>>(y, stats, gamma, beta);
}
